// MultiHeadAttention_82317343195673
// MI455X (gfx1250) — compile-verified, matrix operand order checked by reading, not run on hardware
//
#include <hip/hip_runtime.h>
#include <hip/hip_bf16.h>

#define S_LEN  2048
#define DMODEL 1024
#define NHEAD  16
#define DK     64

typedef __attribute__((ext_vector_type(16))) _Float16 v16h;
typedef __attribute__((ext_vector_type(8)))  _Float16 v8h;
typedef __attribute__((ext_vector_type(8)))  float    v8f;
typedef __attribute__((ext_vector_type(4)))  float    v4f;
typedef __attribute__((ext_vector_type(4)))  int      v4i;

__device__ __forceinline__ v8f wmma16(v16h a, v16h b, v8f c) {
  // D = A(16x32 f16) x B(32x16 f16) + C(16x16 f32)
  return __builtin_amdgcn_wmma_f32_16x16x32_f16(false, a, false, b, (short)0, c,
                                                false, false);
}

// Fragment layout (ISA 16-bit A 16x32):
//   halves 0..7  <- K = kk + 8*hi + j
//   halves 8..15 <- K = kk + 16 + 8*hi + j
__device__ __forceinline__ v16h frag_from_f16(const _Float16* __restrict__ row, int kk, int hi) {
  v8h a = *(const v8h*)(row + kk + 8 * hi);
  v8h b = *(const v8h*)(row + kk + 16 + 8 * hi);
  v16h r;
#pragma unroll
  for (int j = 0; j < 8; ++j) { r[j] = a[j]; r[8 + j] = b[j]; }
  return r;
}

// One-shot fp32 -> f16 conversion (8 elements / thread, b128 in, b128 out).
__global__ void __launch_bounds__(256)
cvt_f32_to_f16(const float* __restrict__ src, _Float16* __restrict__ dst, int n8) {
  const int i = blockIdx.x * blockDim.x + threadIdx.x;
  if (i >= n8) return;
  const v4f* p = (const v4f*)(src + (size_t)i * 8);
  v4f x0 = p[0], x1 = p[1];
  v8h o;
#pragma unroll
  for (int j = 0; j < 4; ++j) {
    o[j]     = (_Float16)x0[j];
    o[4 + j] = (_Float16)x1[j];
  }
  *(v8h*)(dst + (size_t)i * 8) = o;
}

// Y = X @ W^T + bias; one wave computes a 64x64 output tile (16 WMMAs / K-step,
// 4x operand reuse of A and B fragments; pure-f16 fragments -> 1 b128/WMMA).
// STORE_MODE 0: f16 head-major [B,H,S,DK]            (Q,K)
// STORE_MODE 1: f16 head-major transposed [B,H,DK,S] (V)
// STORE_MODE 2: f32 row-major [rows, DMODEL]         (final output)
template <int STORE_MODE>
__global__ void __launch_bounds__(32)
proj_gemm(const _Float16* __restrict__ X, const _Float16* __restrict__ W,
          const float* __restrict__ bias, void* __restrict__ Yv) {
  const int lane = threadIdx.x;
  const int n    = lane & 15;
  const int hi   = lane >> 4;
  const int row0 = blockIdx.x * 64;
  const int col0 = blockIdx.y * 64;

  const _Float16* wrow[4];
  const _Float16* xrow[4];
#pragma unroll
  for (int t = 0; t < 4; ++t) {
    wrow[t] = W + (size_t)(col0 + t * 16 + n) * DMODEL;  // B column n = W row e
    xrow[t] = X + (size_t)(row0 + t * 16 + n) * DMODEL;  // A row for this lane
  }

  v8f acc[4][4];
#pragma unroll
  for (int ms = 0; ms < 4; ++ms)
#pragma unroll
    for (int ns = 0; ns < 4; ++ns) acc[ms][ns] = v8f{};

  for (int kk = 0; kk < DMODEL; kk += 32) {
    v16h a[4], b[4];
#pragma unroll
    for (int ms = 0; ms < 4; ++ms) a[ms] = frag_from_f16(xrow[ms], kk, hi);
#pragma unroll
    for (int ns = 0; ns < 4; ++ns) b[ns] = frag_from_f16(wrow[ns], kk, hi);
#pragma unroll
    for (int ms = 0; ms < 4; ++ms)
#pragma unroll
      for (int ns = 0; ns < 4; ++ns)
        acc[ms][ns] = wmma16(a[ms], b[ns], acc[ms][ns]);
  }

  float be[4];
#pragma unroll
  for (int ns = 0; ns < 4; ++ns) be[ns] = bias[col0 + ns * 16 + n];

  if constexpr (STORE_MODE == 0) {           // f16 [B,H,S,DK]; col tile == one head
    _Float16* Y = (_Float16*)Yv;
    const int head = col0 >> 6;
#pragma unroll
    for (int ms = 0; ms < 4; ++ms) {
#pragma unroll
      for (int ns = 0; ns < 4; ++ns) {
        const int d = ns * 16 + n;
#pragma unroll
        for (int r = 0; r < 8; ++r) {
          const int t  = row0 + ms * 16 + 8 * hi + r;
          const int bb = t >> 11, s = t & (S_LEN - 1);
          Y[(((size_t)bb * NHEAD + head) * S_LEN + s) * DK + d] =
              (_Float16)(acc[ms][ns][r] + be[ns]);
        }
      }
    }
  } else if constexpr (STORE_MODE == 1) {    // f16 [B,H,DK,S]
    _Float16* Y = (_Float16*)Yv;
    const int head = col0 >> 6;
    const int bb   = row0 >> 11;
#pragma unroll
    for (int ms = 0; ms < 4; ++ms) {
      const int s0 = (row0 & (S_LEN - 1)) + ms * 16 + 8 * hi;
#pragma unroll
      for (int ns = 0; ns < 4; ++ns) {
        const int d = ns * 16 + n;
        v8h o;
#pragma unroll
        for (int r = 0; r < 8; ++r) o[r] = (_Float16)(acc[ms][ns][r] + be[ns]);
        *(v8h*)(Y + (((size_t)bb * NHEAD + head) * DK + d) * S_LEN + s0) = o;
      }
    }
  } else {                                   // f32 [rows, DMODEL]
    float* Y = (float*)Yv;
#pragma unroll
    for (int ms = 0; ms < 4; ++ms)
#pragma unroll
      for (int ns = 0; ns < 4; ++ns) {
        const int e = col0 + ns * 16 + n;
#pragma unroll
        for (int r = 0; r < 8; ++r)
          Y[(size_t)(row0 + ms * 16 + 8 * hi + r) * DMODEL + e] = acc[ms][ns][r] + be[ns];
      }
  }
}

// Online softmax over a 32-kv chunk of one q-tile (S^T layout: lane owns one q
// column). Rescales the 4 ctx^T accumulators, returns the f16 P^T B-fragment.
__device__ __forceinline__ v16h online_softmax(v8f s0, v8f s1,
                                               v4i m0a, v4i m0b, v4i m1a, v4i m1b,
                                               float& mrun, float& lrun, v8f* acc) {
  const float scale = 0.125f;  // 1/sqrt(64)
  float cmax = -1e30f;
#pragma unroll
  for (int r = 0; r < 8; ++r) {
    const int mv0 = (r < 4) ? m0a[r] : m0b[r - 4];
    const int mv1 = (r < 4) ? m1a[r] : m1b[r - 4];
    s0[r] = mv0 ? s0[r] * scale : -1e9f;
    s1[r] = mv1 ? s1[r] * scale : -1e9f;
    cmax = fmaxf(cmax, fmaxf(s0[r], s1[r]));
  }
  cmax = fmaxf(cmax, __shfl_xor(cmax, 16, 32));  // combine hi halves of column

  const float mnew  = fmaxf(mrun, cmax);
  const float alpha = __expf(mrun - mnew);
  float lsum = 0.f;
  v16h pb;  // halves j -> kv K = 8*hi+j (tile0) / 16+8*hi+j (tile1)
#pragma unroll
  for (int r = 0; r < 8; ++r) {
    const float p0 = __expf(s0[r] - mnew);
    const float p1 = __expf(s1[r] - mnew);
    lsum += p0 + p1;
    pb[r]     = (_Float16)p0;
    pb[8 + r] = (_Float16)p1;
  }
  lsum += __shfl_xor(lsum, 16, 32);
  lrun = lrun * alpha + lsum;
  mrun = mnew;
#pragma unroll
  for (int dt = 0; dt < 4; ++dt)
#pragma unroll
    for (int r = 0; r < 8; ++r) acc[dt][r] *= alpha;
  return pb;
}

// Flash-style attention; each wave handles TWO 16-wide q-tiles of one (b,h) so
// every K/V fragment feeds two WMMAs. S^T = K.Q^T keeps softmax lane-local.
__global__ void __launch_bounds__(128)
attn_kernel(const _Float16* __restrict__ qh, const _Float16* __restrict__ kh,
            const _Float16* __restrict__ vT, const int* __restrict__ mask,
            _Float16* __restrict__ cc) {
  const int lane = threadIdx.x;
  const int n    = lane & 15;
  const int hi   = lane >> 4;
  const int bh   = blockIdx.x;               // b*NHEAD + h
  const int b    = bh >> 4;
  const int h    = bh & 15;
  const int q0   = (blockIdx.y * 4 + threadIdx.y) * 32;
  const int qc0  = q0 + n;                   // q column, tile A
  const int qc1  = q0 + 16 + n;              // q column, tile B

  const _Float16* qr0 = qh + ((size_t)bh * S_LEN + qc0) * DK;
  const _Float16* qr1 = qh + ((size_t)bh * S_LEN + qc1) * DK;
  const v16h bqA0 = frag_from_f16(qr0, 0, hi), bqA1 = frag_from_f16(qr0, 32, hi);
  const v16h bqB0 = frag_from_f16(qr1, 0, hi), bqB1 = frag_from_f16(qr1, 32, hi);

  const _Float16* kbase = kh + (size_t)bh * S_LEN * DK + (size_t)n * DK;
  const _Float16* vbase = vT + (size_t)bh * DK * S_LEN + (size_t)n * S_LEN;
  const int* mrow0 = mask + ((size_t)b * S_LEN + qc0) * S_LEN;
  const int* mrow1 = mask + ((size_t)b * S_LEN + qc1) * S_LEN;

  float mA = -1e30f, lA = 0.f, mB = -1e30f, lB = 0.f;
  v8f accA[4] = {v8f{}, v8f{}, v8f{}, v8f{}};
  v8f accB[4] = {v8f{}, v8f{}, v8f{}, v8f{}};

  for (int kv0 = 0; kv0 < S_LEN; kv0 += 32) {
    // K fragments for the two 16-kv row tiles, loaded once, used 2x
    const _Float16* kr = kbase + (size_t)kv0 * DK;
    const v16h ka0 = frag_from_f16(kr, 0, hi);
    const v16h ka1 = frag_from_f16(kr, 32, hi);
    const v16h kb0 = frag_from_f16(kr + (size_t)16 * DK, 0, hi);
    const v16h kb1 = frag_from_f16(kr + (size_t)16 * DK, 32, hi);

    v8f sA0 = {}, sA1 = {}, sB0 = {}, sB1 = {};
    sA0 = wmma16(ka0, bqA0, sA0); sA0 = wmma16(ka1, bqA1, sA0);
    sA1 = wmma16(kb0, bqA0, sA1); sA1 = wmma16(kb1, bqA1, sA1);
    sB0 = wmma16(ka0, bqB0, sB0); sB0 = wmma16(ka1, bqB1, sB0);
    sB1 = wmma16(kb0, bqB0, sB1); sB1 = wmma16(kb1, bqB1, sB1);

    const v4i* mpA0 = (const v4i*)(mrow0 + kv0 + 8 * hi);
    const v4i* mpA1 = (const v4i*)(mrow0 + kv0 + 16 + 8 * hi);
    const v4i* mpB0 = (const v4i*)(mrow1 + kv0 + 8 * hi);
    const v4i* mpB1 = (const v4i*)(mrow1 + kv0 + 16 + 8 * hi);

    const v16h pbA = online_softmax(sA0, sA1, mpA0[0], mpA0[1], mpA1[0], mpA1[1],
                                    mA, lA, accA);
    const v16h pbB = online_softmax(sB0, sB1, mpB0[0], mpB0[1], mpB1[0], mpB1[1],
                                    mB, lB, accB);

    // ctx^T[d][q] += sum_kv V^T[d][kv] * P^T[kv][q]; each V fragment used twice
#pragma unroll
    for (int dt = 0; dt < 4; ++dt) {
      const v16h av = frag_from_f16(vbase + (size_t)dt * 16 * S_LEN, kv0, hi);
      accA[dt] = wmma16(av, pbA, accA[dt]);
      accB[dt] = wmma16(av, pbB, accB[dt]);
    }
  }

  const float invA = 1.0f / lA;
  const float invB = 1.0f / lB;
  _Float16* crow0 = cc + ((size_t)b * S_LEN + qc0) * DMODEL + h * DK;
  _Float16* crow1 = cc + ((size_t)b * S_LEN + qc1) * DMODEL + h * DK;
#pragma unroll
  for (int dt = 0; dt < 4; ++dt) {
    v8h oA, oB;
#pragma unroll
    for (int r = 0; r < 8; ++r) {
      oA[r] = (_Float16)(accA[dt][r] * invA);
      oB[r] = (_Float16)(accB[dt][r] * invB);
    }
    *(v8h*)(crow0 + dt * 16 + 8 * hi) = oA;  // d = dt*16 + 8*hi + r, contiguous
    *(v8h*)(crow1 + dt * 16 + 8 * hi) = oB;
  }
}

extern "C" void kernel_launch(void* const* d_in, const int* in_sizes, int n_in,
                              void* d_out, int out_size, void* d_ws, size_t ws_size,
                              hipStream_t stream) {
  const float* q    = (const float*)d_in[0];
  const float* k    = (const float*)d_in[1];
  const float* v    = (const float*)d_in[2];
  const int*   mask = (const int*)d_in[3];
  const float* Wq   = (const float*)d_in[4];
  const float* bq   = (const float*)d_in[5];
  const float* Wk   = (const float*)d_in[6];
  const float* bk   = (const float*)d_in[7];
  const float* Wv   = (const float*)d_in[8];
  const float* bv   = (const float*)d_in[9];
  const float* Wo   = (const float*)d_in[10];
  const float* bo   = (const float*)d_in[11];

  const size_t NN = (size_t)4 * S_LEN * DMODEL;     // 8M f16 elems (activations)
  const size_t NW = (size_t)DMODEL * DMODEL;        // 1M f16 elems (one weight)

  _Float16* q16  = (_Float16*)d_ws;
  _Float16* k16  = q16 + NN;
  _Float16* v16  = k16 + NN;
  _Float16* wq16 = v16 + NN;
  _Float16* wk16 = wq16 + NW;
  _Float16* wv16 = wk16 + NW;
  _Float16* wo16 = wv16 + NW;
  _Float16* qhp  = wo16 + NW;
  _Float16* khp  = qhp + NN;
  _Float16* vTp  = khp + NN;
  _Float16* ccp  = vTp + NN;

  // One-time f32 -> f16 conversion of activations and weights.
  const int actN8 = (int)(NN / 8), wN8 = (int)(NW / 8);
  cvt_f32_to_f16<<<actN8 / 256, 256, 0, stream>>>(q, q16, actN8);
  cvt_f32_to_f16<<<actN8 / 256, 256, 0, stream>>>(k, k16, actN8);
  cvt_f32_to_f16<<<actN8 / 256, 256, 0, stream>>>(v, v16, actN8);
  cvt_f32_to_f16<<<wN8 / 256, 256, 0, stream>>>(Wq, wq16, wN8);
  cvt_f32_to_f16<<<wN8 / 256, 256, 0, stream>>>(Wk, wk16, wN8);
  cvt_f32_to_f16<<<wN8 / 256, 256, 0, stream>>>(Wv, wv16, wN8);
  cvt_f32_to_f16<<<wN8 / 256, 256, 0, stream>>>(Wo, wo16, wN8);

  const dim3 gblk(32, 1);    // one wave, one 64x64 tile
  const dim3 ggrd(128, 16);  // 128 row-tiles x 16 col-tiles

  proj_gemm<0><<<ggrd, gblk, 0, stream>>>(q16, wq16, bq, qhp);
  proj_gemm<0><<<ggrd, gblk, 0, stream>>>(k16, wk16, bk, khp);
  proj_gemm<1><<<ggrd, gblk, 0, stream>>>(v16, wv16, bv, vTp);

  attn_kernel<<<dim3(64, 16), dim3(32, 4), 0, stream>>>(qhp, khp, vTp, mask, ccp);

  proj_gemm<2><<<ggrd, gblk, 0, stream>>>(ccp, wo16, bo, d_out);
}
